// SpecificProfile_28174985462066
// MI455X (gfx1250) — matrix operand order repaired and checked by reading, not compiled, not run on hardware
//
#include <hip/hip_runtime.h>
#include <hip/hip_bf16.h>

typedef __attribute__((ext_vector_type(16))) _Float16 v16h;
typedef __attribute__((ext_vector_type(8)))  float    v8f;

#define T_    4
#define N_    16
#define F_    6
#define L_    1024
#define A_    21
#define K_    20
#define U_    256
#define LOUT  1005           // L - K + 1
#define IMG   384            // T*N*F
#define KDIM  420            // K*A
#define KPAD  448            // 14 chunks of 32
#define NCHUNK 14
#define LTILES 63            // ceil(1005/16)

#define S_SIZE (T_*N_*U_)    // 16384 floats
#define R_SIZE (K_*A_*U_)    // 107520 floats

#define BPERM_HALFS (KPAD*U_)            // 114688 halves = 229376 B
#define SPART_OFF_BYTES (BPERM_HALFS*2)  // Spart: IMG*U_ floats after Bperm

// ---------------------------------------------------------------------------
// Kernel 1: softmax over A per (k,u); R = log(max(P/Q[a], mean(Q)))  (fp32)
// ---------------------------------------------------------------------------
__global__ void prep_softmax_R(const float* __restrict__ Plogit,
                               const float* __restrict__ Q,
                               float* __restrict__ R) {
    int tid = blockIdx.x * blockDim.x + threadIdx.x;   // 0 .. K_*U_-1
    int k = tid / U_;
    int u = tid - k * U_;

    float eps = 0.f;
#pragma unroll
    for (int a = 0; a < A_; ++a) eps += Q[a];
    eps *= (1.0f / (float)A_);

    const float* col = Plogit + (size_t)(k * A_) * U_ + u;
    float vals[A_];
    float m = -INFINITY;
#pragma unroll
    for (int a = 0; a < A_; ++a) { vals[a] = col[(size_t)a * U_]; m = fmaxf(m, vals[a]); }
    float s = 0.f;
#pragma unroll
    for (int a = 0; a < A_; ++a) { vals[a] = __expf(vals[a] - m); s += vals[a]; }
    float inv = 1.0f / s;
#pragma unroll
    for (int a = 0; a < A_; ++a) {
        float p = vals[a] * inv;
        R[(size_t)(k * A_ + a) * U_ + u] = __logf(fmaxf(p / Q[a], eps));
    }
}

// ---------------------------------------------------------------------------
// Kernel 2: pre-swizzle R -> f16 B fragments (32x16 f16 B, contiguous k order:
//           k = e + (lane>=16 ? 16 : 0), n = lane&15), K padded 420 -> 448.
//           Bperm[((kc*16 + nt)*32 + lane)*16 + e]
// ---------------------------------------------------------------------------
__global__ void build_bperm(const float* __restrict__ R,
                            _Float16* __restrict__ Bperm) {
    int tid = blockIdx.x * blockDim.x + threadIdx.x;   // 0 .. NCHUNK*16*32*16-1
    int e    = tid & 15;
    int lane = (tid >> 4) & 31;
    int nt   = (tid >> 9) & 15;
    int kc   = tid >> 13;
    int kk = kc * 32 + e + ((lane >= 16) ? 16 : 0);
    int n  = nt * 16 + (lane & 15);
    float v = (kk < KDIM) ? R[(size_t)kk * U_ + n] : 0.f;
    Bperm[tid] = (_Float16)v;
}

// ---------------------------------------------------------------------------
// Kernel 3: implicit-GEMM conv via v_wmma_f32_16x16x32_f16.
//   Phase 1: coalesced load of 35x21 fp32 X window -> LDS.
//   Phase 2: build swizzled f16 A-fragments for ALL 14 k-chunks ONCE per
//            block (shared by all 8 waves) -> LDS.
//   Phase 3: hot loop = 1 aligned 32B ds load (A) + 2 contiguous 32B global
//            loads (B) + 2 WMMAs per chunk.  B prefetched one chunk ahead.
// ---------------------------------------------------------------------------
__global__ __launch_bounds__(256)
void conv_wmma(const float* __restrict__ X,
               const _Float16* __restrict__ Bperm,
               float* __restrict__ Z) {
    __shared__ float    Xs[35 * A_];            // 735 floats
    __shared__ _Float16 Afrag[NCHUNK * 32 * 16]; // 7168 halves = 14336 B

    const int ltile = blockIdx.x;
    const int img   = blockIdx.y;
    const int l0    = ltile * 16;
    const int tid   = threadIdx.x;

    // Phase 1: X window -> LDS (coalesced)
    for (int idx = tid; idx < 35 * A_; idx += 256) {
        int i  = idx / A_;
        int li = l0 + i;
        Xs[idx] = (li < L_) ? X[(size_t)(img * L_ + li) * A_ + (idx - i * A_)] : 0.f;
    }
    __syncthreads();

    // Phase 2: build swizzled A fragments once per block.
    // Element (kc, lane, e) lives at Afrag[(kc*32+lane)*16 + e].
    // A 16x32, contiguous k order (as in Bperm): k = e + (lane>=16?16:0); im2col
    // collapses to Xs[r*21 + kk] with r = lane&15, kk = kc*32 + k.
    for (int idx = tid; idx < NCHUNK * 32 * 16; idx += 256) {
        int e    = idx & 15;
        int lane = (idx >> 4) & 31;
        int kc   = idx >> 9;
        int r    = lane & 15;
        int kk   = kc * 32 + e + ((lane >= 16) ? 16 : 0);
        float v  = (kk < KDIM) ? Xs[r * A_ + kk] : 0.f;
        Afrag[idx] = (_Float16)v;
    }
    __syncthreads();

    const int lane = tid & 31;
    const int wave = tid >> 5;
    const int nt0  = wave * 2;
    const int nt1  = nt0 + 1;

    const _Float16* bp0 = Bperm + ((size_t)(nt0 * 32 + lane)) * 16;  // +kc*16*32*16
    const _Float16* bp1 = Bperm + ((size_t)(nt1 * 32 + lane)) * 16;
    const int bstep = 16 * 32 * 16;   // halves per k-chunk of Bperm

    v8f c0 = {0.f, 0.f, 0.f, 0.f, 0.f, 0.f, 0.f, 0.f};
    v8f c1 = c0;

    for (int kc = 0; kc < NCHUNK; ++kc) {
        const v16h a  = *(const v16h*)(Afrag + (kc * 32 + lane) * 16);
        const v16h b0 = *(const v16h*)(bp0 + (size_t)kc * bstep);
        const v16h b1 = *(const v16h*)(bp1 + (size_t)kc * bstep);
        if (kc + 1 < NCHUNK) {   // gfx1250 global_prefetch of next chunk's B
            __builtin_prefetch(bp0 + (size_t)(kc + 1) * bstep, 0, 3);
            __builtin_prefetch(bp1 + (size_t)(kc + 1) * bstep, 0, 3);
        }
        c0 = __builtin_amdgcn_wmma_f32_16x16x32_f16(false, a, false, b0, (short)0, c0, false, false);
        c1 = __builtin_amdgcn_wmma_f32_16x16x32_f16(false, a, false, b1, (short)0, c1, false, false);
    }

    // C/D 16x16 f32 layout: VGPR v -> M=v (lanes 0-15) / M=v+8 (lanes 16-31)
    const int col   = lane & 15;
    const int rbase = (lane >= 16) ? 8 : 0;
    float* dst0 = Z + ((size_t)(img * LOUT + l0 + rbase)) * U_ + nt0 * 16 + col;
    float* dst1 = dst0 + 16;   // nt1 tile is +16 columns

    if (l0 + 15 < LOUT) {      // fast path: full tile (62 of 63 tiles)
#pragma unroll
        for (int v = 0; v < 8; ++v) {
            dst0[(size_t)v * U_] = c0[v];
            dst1[(size_t)v * U_] = c1[v];
        }
    } else {
#pragma unroll
        for (int v = 0; v < 8; ++v) {
            if (l0 + rbase + v < LOUT) {
                dst0[(size_t)v * U_] = c0[v];
                dst1[(size_t)v * U_] = c1[v];
            }
        }
    }
}

// ---------------------------------------------------------------------------
// Kernel 4: per-image max over L_out (streaming, coalesced over u)
// ---------------------------------------------------------------------------
__global__ void reduce_img_max(const float* __restrict__ Z,
                               float* __restrict__ Spart) {
    int img = blockIdx.x;
    int u   = threadIdx.x;
    const float* p = Z + (size_t)img * LOUT * U_ + u;
    float m = -INFINITY;
    int l = 0;
    for (; l + 3 < LOUT; l += 4) {
        float a0 = p[(size_t)(l + 0) * U_];
        float a1 = p[(size_t)(l + 1) * U_];
        float a2 = p[(size_t)(l + 2) * U_];
        float a3 = p[(size_t)(l + 3) * U_];
        m = fmaxf(m, fmaxf(fmaxf(a0, a1), fmaxf(a2, a3)));
    }
    for (; l < LOUT; ++l) m = fmaxf(m, p[(size_t)l * U_]);
    Spart[(size_t)img * U_ + u] = m;
}

// ---------------------------------------------------------------------------
// Kernel 5: max over F -> S
// ---------------------------------------------------------------------------
__global__ void reduce_final_max(const float* __restrict__ Spart,
                                 float* __restrict__ S) {
    int tid = blockIdx.x * blockDim.x + threadIdx.x;  // 0 .. S_SIZE-1
    int tn = tid / U_;
    int u  = tid - tn * U_;
    float m = -INFINITY;
#pragma unroll
    for (int f = 0; f < F_; ++f)
        m = fmaxf(m, Spart[(size_t)(tn * F_ + f) * U_ + u]);
    S[tid] = m;
}

// ---------------------------------------------------------------------------
extern "C" void kernel_launch(void* const* d_in, const int* in_sizes, int n_in,
                              void* d_out, int out_size, void* d_ws, size_t ws_size,
                              hipStream_t stream) {
    const float* X      = (const float*)d_in[0];  // (4,16,6,1024,21)
    const float* Plogit = (const float*)d_in[1];  // (20,21,256)
    const float* Q      = (const float*)d_in[2];  // (21,)

    float* out = (float*)d_out;
    float* S = out;                       // 16384
    float* R = out + S_SIZE;              // 107520
    float* Z = out + S_SIZE + R_SIZE;     // 384*1005*256

    _Float16* Bperm = (_Float16*)d_ws;
    float*    Spart = (float*)((char*)d_ws + SPART_OFF_BYTES);

    prep_softmax_R<<<(K_ * U_) / 256, 256, 0, stream>>>(Plogit, Q, R);
    build_bperm<<<(NCHUNK * 16 * 32 * 16) / 256, 256, 0, stream>>>(R, Bperm);
    conv_wmma<<<dim3(LTILES, IMG), 256, 0, stream>>>(X, Bperm, Z);
    reduce_img_max<<<IMG, U_, 0, stream>>>(Z, Spart);
    reduce_final_max<<<S_SIZE / 256, 256, 0, stream>>>(Spart, S);
}
